// MixtureFFNDown_24489903522180
// MI455X (gfx1250) — matrix operand order repaired and checked by reading, not compiled, not run on hardware
//
#include <hip/hip_runtime.h>

typedef __attribute__((ext_vector_type(16))) __bf16 v16bf;
typedef __attribute__((ext_vector_type(8)))  __bf16 v8bf;
typedef __attribute__((ext_vector_type(4)))  __bf16 v4bf;
typedef __attribute__((ext_vector_type(8)))  float  v8f;
typedef __attribute__((ext_vector_type(4)))  int    v4i;

constexpr int B_ = 4, S_ = 2048, H_ = 2048;
constexpr int G_ = 4, E_ = 8, D_ = 512;
constexpr int T_ = B_ * S_;     // 8192 tokens
constexpr int GE = G_ * E_;     // 32 (g,e) pairs

#ifndef __has_builtin
#define __has_builtin(x) 0
#endif
#if __has_builtin(__builtin_amdgcn_global_load_async_to_lds_b128)
#define ASYNC_LDS 1
#else
#define ASYNC_LDS 0
#endif
#if __has_builtin(__builtin_amdgcn_s_wait_asynccnt)
#define HAVE_WAIT_ASYNC 1
#else
#define HAVE_WAIT_ASYNC 0
#endif

// 16-byte global -> LDS copy: async (ASYNCcnt) when available, sync fallback.
__device__ __forceinline__ void copy16_g2l(const __bf16* g, __bf16* l) {
#if ASYNC_LDS
    __builtin_amdgcn_global_load_async_to_lds_b128(
        (__attribute__((address_space(1))) v4i*)(uintptr_t)g,
        (__attribute__((address_space(3))) v4i*)(unsigned)(uintptr_t)l,
        0, 0);
#else
    *reinterpret_cast<v8bf*>(l) = *reinterpret_cast<const v8bf*>(g);
#endif
}

__device__ __forceinline__ void wait_async_zero() {
#if HAVE_WAIT_ASYNC
    __builtin_amdgcn_s_wait_asynccnt(0);
#elif ASYNC_LDS
    asm volatile("s_wait_asynccnt 0x0" ::: "memory");
#endif
}

__device__ __forceinline__ v16bf cat8(v8bf lo, v8bf hi) {
    return __builtin_shufflevector(lo, hi, 0,1,2,3,4,5,6,7,8,9,10,11,12,13,14,15);
}

// ---------------------------------------------------------------------------
// Main MoE GEMM with double-buffered async LDS staging. (First in module so
// the disasm snippet shows its inner loop.)
// grid = (T/64, D/256, G); block = 256 = 8 waves (2 m-waves x 4 n-waves).
// Block tile per K=32 step: A 64x32 bf16 (4KB), B 256x32 bf16 (16KB),
// rows padded to 80B for conflict-free ds_load_b128. 512 flat steps
// (8 experts x 64 k-steps); epilogue folds combine*(c+bias) per expert.
// ---------------------------------------------------------------------------
constexpr int LDA = 40;                 // bf16 elements per LDS row (80 B)
constexpr int STEPS_PER_E = H_ / 32;    // 64
constexpr int TOTAL_STEPS = E_ * STEPS_PER_E;

__global__ __launch_bounds__(256)
void moe_gemm_kernel(const __bf16* __restrict__ Xb, const __bf16* __restrict__ Wb,
                     const float* __restrict__ expert_b,
                     const float* __restrict__ combine,
                     float* __restrict__ acc) {
    __shared__ __align__(16) __bf16 ldsA[2][64 * LDA];   // 2 x 5 KB
    __shared__ __align__(16) __bf16 ldsB[2][256 * LDA];  // 2 x 20 KB

    const int tid  = threadIdx.x;
    const int lane = tid & 31;
    const int wave = tid >> 5;
    const int tokenBlock = blockIdx.x * 64;
    const int dBlock     = blockIdx.y * 256;
    const int tokenBase  = tokenBlock + (wave & 1) * 32;
    const int dBase      = dBlock + (wave >> 1) * 64;
    const int g = blockIdx.z;
    const int hi16 = lane >> 4;
    const int lr   = lane & 15;

    // Stage the A(4KB)+B(16KB) tiles for flat step s into buffer `buf`.
    // A: 256 segments of 16B -> one per thread. B: 1024 segments -> 4 per thread.
    auto stage = [&](int s, int buf) {
        const int es = s >> 6;
        const int k0 = (s & 63) * 32;
        {
            int row = tid >> 2, seg = tid & 3;
            copy16_g2l(Xb + (size_t)(tokenBlock + row) * H_ + k0 + seg * 8,
                       &ldsA[buf][row * LDA + seg * 8]);
        }
        const __bf16* wrow = Wb + ((size_t)(g * E_ + es) * D_ + dBlock) * H_;
        #pragma unroll
        for (int p = 0; p < 4; ++p) {
            int j = tid + 256 * p;
            int row = j >> 2, seg = j & 3;
            copy16_g2l(wrow + (size_t)row * H_ + k0 + seg * 8,
                       &ldsB[buf][row * LDA + seg * 8]);
        }
    };

    v8f out[2][4];
    #pragma unroll
    for (int mi = 0; mi < 2; ++mi)
        #pragma unroll
        for (int ni = 0; ni < 4; ++ni) out[mi][ni] = v8f{};

    // LDS fragment base pointers (per wave/lane), constant across steps.
    const __bf16* aBase[2];
    const __bf16* bBase[4];
    #pragma unroll
    for (int mi = 0; mi < 2; ++mi)
        aBase[mi] = &ldsA[0][((wave & 1) * 32 + mi * 16 + lr) * LDA + hi16 * 8];
    #pragma unroll
    for (int ni = 0; ni < 4; ++ni)
        bBase[ni] = &ldsB[0][((wave >> 1) * 64 + ni * 16 + lr) * LDA + hi16 * 8];
    const size_t bufStrideA = 64 * LDA;   // elements between ldsA buffers
    const size_t bufStrideB = 256 * LDA;

    stage(0, 0);                          // prologue

    int step = 0;
    for (int e = 0; e < E_; ++e) {
        v8f c[2][4];
        #pragma unroll
        for (int mi = 0; mi < 2; ++mi)
            #pragma unroll
            for (int ni = 0; ni < 4; ++ni) c[mi][ni] = v8f{};

        for (int ks = 0; ks < STEPS_PER_E; ++ks, ++step) {
            const int buf = step & 1;
            wait_async_zero();            // this wave's staged batch landed
            __syncthreads();              // everyone's batch landed / reads done
            if (step + 1 < TOTAL_STEPS) stage(step + 1, buf ^ 1);

            v16bf A[2], Bm[4];
            #pragma unroll
            for (int mi = 0; mi < 2; ++mi) {
                const __bf16* p = aBase[mi] + buf * bufStrideA;
                A[mi] = cat8(*reinterpret_cast<const v8bf*>(p),
                             *reinterpret_cast<const v8bf*>(p + 16));
            }
            #pragma unroll
            for (int ni = 0; ni < 4; ++ni) {
                const __bf16* p = bBase[ni] + buf * bufStrideB;
                Bm[ni] = cat8(*reinterpret_cast<const v8bf*>(p),
                              *reinterpret_cast<const v8bf*>(p + 16));
            }
            #pragma unroll
            for (int mi = 0; mi < 2; ++mi)
                #pragma unroll
                for (int ni = 0; ni < 4; ++ni)
                    c[mi][ni] = __builtin_amdgcn_wmma_f32_16x16x32_bf16(
                        false, A[mi], false, Bm[ni], (short)0, c[mi][ni], false, false);
        }

        // epilogue for this expert: out += combine * (c + bias)
        float bias[4];
        #pragma unroll
        for (int ni = 0; ni < 4; ++ni)
            bias[ni] = expert_b[(size_t)(g * E_ + e) * D_ + dBase + ni * 16 + lr];
        #pragma unroll
        for (int mi = 0; mi < 2; ++mi) {
            #pragma unroll
            for (int r = 0; r < 8; ++r) {
                int tok = tokenBase + mi * 16 + hi16 * 8 + r;
                float cwv = combine[(size_t)tok * GE + g * E_ + e];
                #pragma unroll
                for (int ni = 0; ni < 4; ++ni)
                    out[mi][ni][r] += cwv * (c[mi][ni][r] + bias[ni]);
            }
        }
    }

    #pragma unroll
    for (int mi = 0; mi < 2; ++mi)
        #pragma unroll
        for (int ni = 0; ni < 4; ++ni)
            #pragma unroll
            for (int r = 0; r < 8; ++r) {
                int tok = tokenBase + mi * 16 + hi16 * 8 + r;
                int d   = dBase + ni * 16 + lr;
                atomicAdd(&acc[(size_t)tok * D_ + d], out[mi][ni][r]);
            }
}

// ---------------------------------------------------------------------------
// f32 -> bf16 conversion, 4 elements per thread
// ---------------------------------------------------------------------------
__global__ void f32_to_bf16x4_kernel(const float* __restrict__ src,
                                     __bf16* __restrict__ dst, int n4) {
    int i = blockIdx.x * blockDim.x + threadIdx.x;
    int stride = gridDim.x * blockDim.x;
    for (; i < n4; i += stride) {
        float4 f = reinterpret_cast<const float4*>(src)[i];
        v4bf o;
        o[0] = (__bf16)f.x; o[1] = (__bf16)f.y;
        o[2] = (__bf16)f.z; o[3] = (__bf16)f.w;
        reinterpret_cast<v4bf*>(dst)[i] = o;
    }
}

// ---------------------------------------------------------------------------
// Gate GEMM: logits[T][32] = Xb[T][H] * Gb[32][H]^T (small: direct from global)
// ---------------------------------------------------------------------------
__global__ __launch_bounds__(256)
void gate_gemm_kernel(const __bf16* __restrict__ Xb, const __bf16* __restrict__ Gb,
                      float* __restrict__ logits) {
    const int lane = threadIdx.x & 31;
    const int wave = threadIdx.x >> 5;
    const int tokenBase = (blockIdx.x * 8 + wave) * 16;
    const int hi16 = lane >> 4;
    const int lr   = lane & 15;

    const __bf16* aP  = Xb + (size_t)(tokenBase + lr) * H_ + hi16 * 8;
    const __bf16* bP0 = Gb + (size_t)(lr)      * H_ + hi16 * 8;
    const __bf16* bP1 = Gb + (size_t)(16 + lr) * H_ + hi16 * 8;

    v8f c0 = {}; v8f c1 = {};
    for (int k0 = 0; k0 < H_; k0 += 32) {
        v16bf a  = cat8(*reinterpret_cast<const v8bf*>(aP  + k0),
                        *reinterpret_cast<const v8bf*>(aP  + k0 + 16));
        v16bf b0 = cat8(*reinterpret_cast<const v8bf*>(bP0 + k0),
                        *reinterpret_cast<const v8bf*>(bP0 + k0 + 16));
        v16bf b1 = cat8(*reinterpret_cast<const v8bf*>(bP1 + k0),
                        *reinterpret_cast<const v8bf*>(bP1 + k0 + 16));
        c0 = __builtin_amdgcn_wmma_f32_16x16x32_bf16(false, a, false, b0, (short)0, c0, false, false);
        c1 = __builtin_amdgcn_wmma_f32_16x16x32_bf16(false, a, false, b1, (short)0, c1, false, false);
    }
    #pragma unroll
    for (int r = 0; r < 8; ++r) {
        int tok = tokenBase + hi16 * 8 + r;
        logits[(size_t)tok * GE + lr]      = c0[r];
        logits[(size_t)tok * GE + 16 + lr] = c1[r];
    }
}

// ---------------------------------------------------------------------------
// Routing: softmax over E=8, top-2, renormalize. One thread per (t, g).
// ---------------------------------------------------------------------------
__global__ void routing_kernel(const float* __restrict__ logits,
                               float* __restrict__ combine) {
    int idx = blockIdx.x * blockDim.x + threadIdx.x;
    if (idx >= T_ * G_) return;
    int t = idx >> 2, g = idx & 3;
    const float* lg = logits + (size_t)t * GE + g * E_;
    float l[E_]; float m = -3.4e38f;
    #pragma unroll
    for (int e = 0; e < E_; ++e) { l[e] = lg[e]; m = fmaxf(m, l[e]); }
    float p[E_];
    #pragma unroll
    for (int e = 0; e < E_; ++e) p[e] = __expf(l[e] - m);
    int i1 = 0; float v1 = p[0];
    #pragma unroll
    for (int e = 1; e < E_; ++e) if (p[e] > v1) { v1 = p[e]; i1 = e; }
    int i2 = -1; float v2 = -1.f;
    #pragma unroll
    for (int e = 0; e < E_; ++e) if (e != i1 && p[e] > v2) { v2 = p[e]; i2 = e; }
    float inv = 1.f / (v1 + v2);
    float* cw = combine + (size_t)t * GE + g * E_;
    #pragma unroll
    for (int e = 0; e < E_; ++e)
        cw[e] = (e == i1) ? v1 * inv : ((e == i2) ? v2 * inv : 0.f);
}

// ---------------------------------------------------------------------------
// Final: out[t,g2] = x[t].orig_w[g2] + orig_b[g2]
//                  + (1/G) * sum_d acc[t,d]*agg_w[g2,d] + agg_b[g2]
// ---------------------------------------------------------------------------
__global__ __launch_bounds__(256)
void final_kernel(const float* __restrict__ x, const float* __restrict__ acc,
                  const float* __restrict__ agg_w, const float* __restrict__ agg_b,
                  const float* __restrict__ orig_w, const float* __restrict__ orig_b,
                  float* __restrict__ outp) {
    const int lane = threadIdx.x & 31;
    const int wave = threadIdx.x >> 5;
    const int t = blockIdx.x * 8 + wave;

    float aggS[G_]  = {0.f, 0.f, 0.f, 0.f};
    float origS[G_] = {0.f, 0.f, 0.f, 0.f};

    const float* ac = acc + (size_t)t * D_;
    for (int d = lane; d < D_; d += 32) {
        float v = ac[d];
        #pragma unroll
        for (int g2 = 0; g2 < G_; ++g2) aggS[g2] += v * agg_w[g2 * D_ + d];
    }
    const float* xr = x + (size_t)t * H_;
    for (int h = lane; h < H_; h += 32) {
        float v = xr[h];
        #pragma unroll
        for (int g2 = 0; g2 < G_; ++g2) origS[g2] += v * orig_w[g2 * H_ + h];
    }
    #pragma unroll
    for (int off = 16; off > 0; off >>= 1) {
        #pragma unroll
        for (int g2 = 0; g2 < G_; ++g2) {
            aggS[g2]  += __shfl_xor(aggS[g2],  off, 32);
            origS[g2] += __shfl_xor(origS[g2], off, 32);
        }
    }
    if (lane == 0) {
        #pragma unroll
        for (int g2 = 0; g2 < G_; ++g2)
            outp[(size_t)t * G_ + g2] =
                origS[g2] + orig_b[g2] + aggS[g2] * (1.f / G_) + agg_b[g2];
    }
}

// ---------------------------------------------------------------------------
extern "C" void kernel_launch(void* const* d_in, const int* in_sizes, int n_in,
                              void* d_out, int out_size, void* d_ws, size_t ws_size,
                              hipStream_t stream) {
    (void)in_sizes; (void)n_in; (void)out_size; (void)ws_size;
    const float* x        = (const float*)d_in[0];
    const float* gate_w   = (const float*)d_in[1];
    const float* expert_w = (const float*)d_in[2];
    const float* expert_b = (const float*)d_in[3];
    const float* agg_w    = (const float*)d_in[4];
    const float* agg_b    = (const float*)d_in[5];
    const float* orig_w   = (const float*)d_in[6];
    const float* orig_b   = (const float*)d_in[7];
    float* outp = (float*)d_out;

    char* ws = (char*)d_ws;
    size_t off = 0;
    auto wsAlloc = [&](size_t bytes) -> void* {
        void* p = (void*)(ws + off);
        off += (bytes + 255) & ~(size_t)255;
        return p;
    };
    __bf16* xb      = (__bf16*)wsAlloc((size_t)T_ * H_ * 2);        // 32 MB
    __bf16* wb      = (__bf16*)wsAlloc((size_t)GE * D_ * H_ * 2);   // 64 MB
    __bf16* gb      = (__bf16*)wsAlloc((size_t)GE * H_ * 2);        // 128 KB
    float*  logits  = (float*)wsAlloc((size_t)T_ * GE * 4);         // 1 MB
    float*  combine = (float*)wsAlloc((size_t)T_ * GE * 4);         // 1 MB
    float*  acc     = (float*)wsAlloc((size_t)T_ * D_ * 4);         // 16 MB

    {
        int n4 = T_ * H_ / 4;
        f32_to_bf16x4_kernel<<<(n4 + 255) / 256, 256, 0, stream>>>(x, xb, n4);
    }
    {
        int n4 = GE * D_ * H_ / 4;
        f32_to_bf16x4_kernel<<<(n4 + 255) / 256, 256, 0, stream>>>(expert_w, wb, n4);
    }
    {
        int n4 = GE * H_ / 4;
        f32_to_bf16x4_kernel<<<(n4 + 255) / 256, 256, 0, stream>>>(gate_w, gb, n4);
    }

    (void)hipMemsetAsync(acc, 0, (size_t)T_ * D_ * 4, stream);

    gate_gemm_kernel<<<T_ / 128, 256, 0, stream>>>(xb, gb, logits);
    routing_kernel<<<(T_ * G_ + 255) / 256, 256, 0, stream>>>(logits, combine);

    dim3 grid(T_ / 64, D_ / 256, G_);
    moe_gemm_kernel<<<grid, 256, 0, stream>>>(xb, wb, expert_b, combine, acc);

    final_kernel<<<T_ / 8, 256, 0, stream>>>(x, acc, agg_w, agg_b, orig_w, orig_b, outp);
}
